// MultiHeadAttention_56075093016741
// MI455X (gfx1250) — matrix operand order repaired and checked by reading, not compiled, not run on hardware
//
#include <hip/hip_runtime.h>

// ---------------- problem constants (from reference) ----------------
constexpr int BATCH = 2;
constexpr int SEQ   = 2048;
constexpr int NHEAD = 16;
constexpr int HEADD = 64;
constexpr int HDIM  = NHEAD * HEADD;   // 1024
constexpr int MTOK  = BATCH * SEQ;     // 4096
constexpr int LDSB  = 40;              // padded LDS row stride (halves) for 32-wide K tiles

typedef __attribute__((ext_vector_type(16))) __bf16 bf16x16;
typedef __attribute__((ext_vector_type(8)))  float  f32x8;

union Frag16 { bf16x16 v; uint4 q[2]; };

#define DEVINL static __device__ __forceinline__

DEVINL unsigned short f2bf(float f) {      // fp32 -> bf16, round-to-nearest-even
  unsigned int u = __float_as_uint(f);
  u += 0x7FFFu + ((u >> 16) & 1u);
  return (unsigned short)(u >> 16);
}

// lane-XOR exchange via ds_swizzle (group-of-32 mode: [14:10]=xor, [4:0]=and)
#define SWZ_XOR(v, m)                                                          \
  __uint_as_float((unsigned)__builtin_amdgcn_ds_swizzle(                       \
      (int)__float_as_uint(v), (((m) << 10) | 0x1F)))

// CDNA5 async global->LDS copy (ASYNCcnt-tracked, no VGPR data path)
DEVINL void async_load_b128(unsigned lds_off, const void* gaddr) {
  asm volatile("global_load_async_to_lds_b128 %0, %1, off"
               :: "v"(lds_off), "v"((unsigned long long)gaddr)
               : "memory");
}
DEVINL void wait_asynccnt0() {
  asm volatile("s_wait_asynccnt 0x0" ::: "memory");
}

// A-matrix fragment (16x32 bf16), ISA 7.12.2: lane l -> row M = l&15,
// K chunks [h*8, h*8+8) and [16+h*8, 16+h*8+8) with h = l>>4.
DEVINL bf16x16 load_frag_a(const unsigned short* base, int ld, int k0, int lane) {
  const unsigned short* p = base + (lane & 15) * ld + k0 + ((lane >> 4) << 3);
  Frag16 f;
  f.q[0] = *(const uint4*)(p);
  f.q[1] = *(const uint4*)(p + 16);
  return f.v;
}

// B-matrix fragment (32x16 bf16) from row-major [N][K] storage:
// lane l -> column n = l&15 (a stored row), K chunks [h*8, h*8+8) and [16+h*8, 16+h*8+8), h = l>>4.
DEVINL bf16x16 load_frag_b(const unsigned short* base, int ld, int k0, int lane) {
  const unsigned short* p = base + (lane & 15) * ld + k0 + ((lane >> 4) << 3);
  Frag16 f;
  f.q[0] = *(const uint4*)(p);
  f.q[1] = *(const uint4*)(p + 16);
  return f.v;
}

DEVINL f32x8 wmma_bf16(bf16x16 a, bf16x16 b, f32x8 c) {
  return __builtin_amdgcn_wmma_f32_16x16x32_bf16(false, a, false, b, (short)0, c,
                                                 false, false);
}

// ---------------- fp32 -> bf16 bulk conversion ----------------
__global__ __launch_bounds__(256)
void cvt_f32_to_bf16(const float* __restrict__ in, unsigned short* __restrict__ out,
                     int n) {
  int i = (blockIdx.x * 256 + threadIdx.x) * 4;
  if (i < n) {
    float4 f = *(const float4*)(in + i);
    uint2 o;
    o.x = (unsigned)f2bf(f.x) | ((unsigned)f2bf(f.y) << 16);
    o.y = (unsigned)f2bf(f.z) | ((unsigned)f2bf(f.w) << 16);
    *(uint2*)(out + i) = o;
  }
}

// ---------------- GEMM: C[M,N] = A[M,K] @ W[N,K]^T + bias ----------------
// M = MTOK, N = K = HDIM.  Block = 8 waves; block tile 128x64; wave tile 16x64.
// W tile (64x32) double-buffered in LDS, filled with async global->LDS copies
// that overlap the WMMAs of the previous K-step.
// OMODE 0: bf16 row-major   1: bf16 per-head transposed [B,NH,HD,T] (for V)
// OMODE 2: fp32 row-major (final output)
template <int OMODE>
__global__ __launch_bounds__(256)
void gemm_bt(const unsigned short* __restrict__ A,
             const unsigned short* __restrict__ W,
             const float* __restrict__ bias,
             void* __restrict__ Cout) {
  __shared__ unsigned short sB[2][64 * LDSB];
  const int lane   = threadIdx.x & 31;
  const int wave   = threadIdx.x >> 5;
  const int tile_n = blockIdx.x * 64;
  const int tile_m = blockIdx.y * 128;

  const f32x8 vzero = {0.f, 0.f, 0.f, 0.f, 0.f, 0.f, 0.f, 0.f};
  f32x8 acc[4] = {vzero, vzero, vzero, vzero};

  const unsigned short* aBase = A + (size_t)(tile_m + wave * 16) * HDIM;

  const int tid  = threadIdx.x;
  const int brow = tid >> 2;          // 0..63
  const int bcol = (tid & 3) * 8;     // 0,8,16,24  (halves)
  const unsigned short* wRow = W + (size_t)(tile_n + brow) * HDIM + bcol;
  const unsigned ldsDst[2] = {
      (unsigned)(size_t)&sB[0][brow * LDSB + bcol],
      (unsigned)(size_t)&sB[1][brow * LDSB + bcol]};

  // prime buffer 0
  async_load_b128(ldsDst[0], wRow);
  wait_asynccnt0();
  __syncthreads();

  for (int kk = 0; kk < HDIM; kk += 32) {
    const int cur = (kk >> 5) & 1;
    if (kk + 32 < HDIM)                      // prefetch next tile into other buffer
      async_load_b128(ldsDst[cur ^ 1], wRow + kk + 32);

    bf16x16 aF = load_frag_a(aBase, HDIM, kk, lane);
#pragma unroll
    for (int f = 0; f < 4; ++f) {
      bf16x16 bF = load_frag_b(&sB[cur][f * 16 * LDSB], LDSB, 0, lane);
      acc[f] = wmma_bf16(aF, bF, acc[f]);
    }

    wait_asynccnt0();                        // next tile landed in LDS
    __syncthreads();                         // all waves done reading cur buffer
  }

  const int halfsel = lane >> 4, col = lane & 15;
#pragma unroll
  for (int f = 0; f < 4; ++f) {
    const int n = tile_n + f * 16 + col;
    const float bval = bias[n];
#pragma unroll
    for (int r = 0; r < 8; ++r) {
      const int m = tile_m + wave * 16 + r + halfsel * 8;  // C layout: M = r + 8*(lane>=16)
      const float v = acc[f][r] + bval;
      if (OMODE == 0) {
        ((unsigned short*)Cout)[(size_t)m * HDIM + n] = f2bf(v);
      } else if (OMODE == 1) {
        const int bb = m >> 11, tt = m & (SEQ - 1);
        const int hh = n >> 6,  dd = n & (HEADD - 1);
        ((unsigned short*)Cout)[((size_t)(bb * NHEAD + hh) * HEADD + dd) * SEQ + tt] =
            f2bf(v);
      } else {
        ((float*)Cout)[(size_t)m * HDIM + n] = v;
      }
    }
  }
}

// ---------------- fused flash attention ----------------
// One wave per (b, h, 16-query tile). Scores never leave registers/LDS.
__global__ __launch_bounds__(32)
void attn_fused(const unsigned short* __restrict__ Q,
                const unsigned short* __restrict__ Km,
                const unsigned short* __restrict__ Vt,   // [B,NH,HD,T] bf16
                const int* __restrict__ valid_len,
                unsigned short* __restrict__ ctx) {
  __shared__ unsigned short sP[16 * LDSB];
  const int lane  = threadIdx.x & 31;
  const int qt    = blockIdx.x;
  const int h     = blockIdx.y;
  const int b     = blockIdx.z;
  const int vlen  = valid_len[b];
  const int qbase = qt * 16;
  const int halfsel = lane >> 4, col = lane & 15;

  const unsigned short* qBase = Q + (size_t)(b * SEQ + qbase) * HDIM + h * HEADD;
  const bf16x16 qF0 = load_frag_a(qBase, HDIM, 0, lane);
  const bf16x16 qF1 = load_frag_a(qBase, HDIM, 32, lane);

  const unsigned short* kBase = Km + (size_t)(b * SEQ) * HDIM + h * HEADD;
  const unsigned short* vBase = Vt + (size_t)(b * NHEAD + h) * HEADD * SEQ;

  float m_r[8], l_r[8];
#pragma unroll
  for (int r = 0; r < 8; ++r) { m_r[r] = -1e30f; l_r[r] = 0.f; }
  const f32x8 vzero = {0.f, 0.f, 0.f, 0.f, 0.f, 0.f, 0.f, 0.f};
  f32x8 acc[4] = {vzero, vzero, vzero, vzero};

  const int nkb = (vlen + 31) >> 5;          // only touch key blocks < valid_len
  for (int kb = 0; kb < nkb; ++kb) {
    const int key0 = kb * 32;
    const unsigned short* kT0 = kBase + (size_t)key0 * HDIM;
    const unsigned short* kT1 = kBase + (size_t)(key0 + 16) * HDIM;

    // S = Q @ K^T for 16x32 keys (two 16x16 tiles, K-dim = HD = 64)
    f32x8 s0 = vzero, s1 = vzero;
    s0 = wmma_bf16(qF0, load_frag_b(kT0, HDIM, 0, lane), s0);
    s0 = wmma_bf16(qF1, load_frag_b(kT0, HDIM, 32, lane), s0);
    s1 = wmma_bf16(qF0, load_frag_b(kT1, HDIM, 0, lane), s1);
    s1 = wmma_bf16(qF1, load_frag_b(kT1, HDIM, 32, lane), s1);

    const bool v0 = (key0 + col) < vlen;          // key-position mask (per column)
    const bool v1 = (key0 + 16 + col) < vlen;

#pragma unroll
    for (int r = 0; r < 8; ++r) {                 // row = r + 8*halfsel
      float a0 = v0 ? s0[r] * 0.125f : -1e30f;    // 1/sqrt(64)
      float a1 = v1 ? s1[r] * 0.125f : -1e30f;
      float rm = fmaxf(a0, a1);                   // row-max across 16 columns
      rm = fmaxf(rm, SWZ_XOR(rm, 1));
      rm = fmaxf(rm, SWZ_XOR(rm, 2));
      rm = fmaxf(rm, SWZ_XOR(rm, 4));
      rm = fmaxf(rm, SWZ_XOR(rm, 8));
      const float mn  = fmaxf(m_r[r], rm);
      const float fac = __expf(m_r[r] - mn);      // exp(-1e30-..)=0 at first block
      const float p0  = __expf(a0 - mn);
      const float p1  = __expf(a1 - mn);
      float rs = p0 + p1;                         // row-sum across 16 columns
      rs += SWZ_XOR(rs, 1);
      rs += SWZ_XOR(rs, 2);
      rs += SWZ_XOR(rs, 4);
      rs += SWZ_XOR(rs, 8);
      l_r[r] = l_r[r] * fac + rs;
      m_r[r] = mn;
#pragma unroll
      for (int f = 0; f < 4; ++f) acc[f][r] *= fac;
      sP[(r + halfsel * 8) * LDSB + col]      = f2bf(p0);
      sP[(r + halfsel * 8) * LDSB + 16 + col] = f2bf(p1);
    }

    asm volatile("s_wait_dscnt 0x0" ::: "memory");  // P stores -> P fragment loads
    const bf16x16 pF = load_frag_a(sP, LDSB, 0, lane);
#pragma unroll
    for (int f = 0; f < 4; ++f) {                 // ctx += P @ V (V^T rows contiguous)
      bf16x16 vF = load_frag_b(vBase + (size_t)(f * 16) * SEQ, SEQ, key0, lane);
      acc[f] = wmma_bf16(pF, vF, acc[f]);
    }
  }

#pragma unroll
  for (int f = 0; f < 4; ++f) {
    const int d = f * 16 + col;
#pragma unroll
    for (int r = 0; r < 8; ++r) {
      const int tq = qbase + r + halfsel * 8;
      const float v = acc[f][r] / l_r[r];
      ctx[(size_t)(b * SEQ + tq) * HDIM + h * HEADD + d] = f2bf(v);
    }
  }
}

// ---------------- host-side launch ----------------
extern "C" void kernel_launch(void* const* d_in, const int* in_sizes, int n_in,
                              void* d_out, int out_size, void* d_ws, size_t ws_size,
                              hipStream_t stream) {
  (void)in_sizes; (void)n_in; (void)out_size; (void)ws_size;
  const float* x    = (const float*)d_in[0];
  const float* Wq   = (const float*)d_in[1];
  const float* bq   = (const float*)d_in[2];
  const float* Wk   = (const float*)d_in[3];
  const float* bk   = (const float*)d_in[4];
  const float* Wv   = (const float*)d_in[5];
  const float* bv   = (const float*)d_in[6];
  const float* Wo   = (const float*)d_in[7];
  const float* bo   = (const float*)d_in[8];
  const int*   vlen = (const int*)d_in[9];

  char* ws = (char*)d_ws;
  const size_t xbB = (size_t)MTOK * HDIM * 2;   // 8 MB per [M,H] bf16 buffer
  const size_t wB  = (size_t)HDIM * HDIM * 2;   // 2 MB per weight
  unsigned short* xb  = (unsigned short*)(ws);
  unsigned short* Wqb = (unsigned short*)(ws + xbB);
  unsigned short* Wkb = (unsigned short*)(ws + xbB + 1 * wB);
  unsigned short* Wvb = (unsigned short*)(ws + xbB + 2 * wB);
  unsigned short* Wob = (unsigned short*)(ws + xbB + 3 * wB);
  unsigned short* Qb  = (unsigned short*)(ws + 1 * xbB + 4 * wB);
  unsigned short* Kb  = (unsigned short*)(ws + 2 * xbB + 4 * wB);
  unsigned short* Vt  = (unsigned short*)(ws + 3 * xbB + 4 * wB);
  unsigned short* Cx  = (unsigned short*)(ws + 4 * xbB + 4 * wB);  // total 48 MB

  const int nX = MTOK * HDIM;   // 4194304
  const int nW = HDIM * HDIM;   // 1048576
  cvt_f32_to_bf16<<<nX / 1024, 256, 0, stream>>>(x, xb, nX);
  cvt_f32_to_bf16<<<nW / 1024, 256, 0, stream>>>(Wq, Wqb, nW);
  cvt_f32_to_bf16<<<nW / 1024, 256, 0, stream>>>(Wk, Wkb, nW);
  cvt_f32_to_bf16<<<nW / 1024, 256, 0, stream>>>(Wv, Wvb, nW);
  cvt_f32_to_bf16<<<nW / 1024, 256, 0, stream>>>(Wo, Wob, nW);

  dim3 ggrid(HDIM / 64, MTOK / 128);
  gemm_bt<0><<<ggrid, 256, 0, stream>>>(xb, Wqb, bq, Qb);
  gemm_bt<0><<<ggrid, 256, 0, stream>>>(xb, Wkb, bk, Kb);
  gemm_bt<1><<<ggrid, 256, 0, stream>>>(xb, Wvb, bv, Vt);   // V stored transposed

  attn_fused<<<dim3(SEQ / 16, NHEAD, BATCH), 32, 0, stream>>>(Qb, Kb, Vt, vlen, Cx);

  gemm_bt<2><<<ggrid, 256, 0, stream>>>(Cx, Wob, bo, d_out);
}
